// HeterogeneousGraphNetworkSimulator_14130442403993
// MI455X (gfx1250) — matrix operand order repaired and checked by reading, not compiled, not run on hardware
//
#include <hip/hip_runtime.h>

#define LAT 64
#define LN_EPS 1e-6f

typedef __attribute__((ext_vector_type(16))) _Float16 v16h;
typedef __attribute__((ext_vector_type(8)))  _Float16 v8h;
typedef __attribute__((ext_vector_type(4)))  _Float16 h4;
typedef __fp16 __attribute__((ext_vector_type(2))) fp16x2;
typedef __attribute__((ext_vector_type(8)))  float    v8f;

__device__ __forceinline__ v8f wmma_f16(v16h a, v16h b, v8f c) {
  // D = A(16x32 f16) * B(32x16 f16) + C(16x16 f32)
  return __builtin_amdgcn_wmma_f32_16x16x32_f16(false, a, false, b, (short)0, c,
                                                false, false);
}

// two contiguous 16B LDS loads -> one v16h fragment
__device__ __forceinline__ v16h ld_v16h(const _Float16* p0,
                                        const _Float16* p1) {
  v8h a = *(const v8h*)p0;
  v8h b = *(const v8h*)p1;
  return __builtin_shufflevector(a, b, 0, 1, 2, 3, 4, 5, 6, 7, 8, 9, 10, 11,
                                 12, 13, 14, 15);
}

// A fragment (16x32 f16) from LDS row-major X[16][ldx], K-offset kb.
// lanes 0-15 row M=lane: K=0..15; lanes 16-31: K=16..31 (matches Wf swizzle).
// Per lane this is two contiguous 16B runs -> 2x ds_load_b128.
__device__ __forceinline__ v16h ld_A(const _Float16* X, int ldx, int kb,
                                     int lo, int h) {
  const _Float16* p = X + lo * ldx + kb + h * 16;
  return ld_v16h(p, p + 8);
}

// B fragment from pre-swizzled weight storage: frag f, lane l own 16
// contiguous halves at (f*32+l)*16 -> 2x ds_load_b128.
__device__ __forceinline__ v16h ld_Bf(const _Float16* Wf, int frag, int lane) {
  const _Float16* p = Wf + (size_t)(frag * 32 + lane) * 16;
  return ld_v16h(p, p + 8);
}

// Swizzle global W[Kreal][64] f32 (zero-pad to Kpad rows) into LDS fragment
// layout: value W[k*32 + h*16 + j][t*16 + lo] -> Wf[((k*4+t)*32 + h*16+lo)*16 + j]
__device__ __forceinline__ void swizzle_W64(const float* __restrict__ W,
                                            _Float16* Wf, int Kpad, int Kreal,
                                            int tid) {
  for (int i = tid; i < Kpad * 64; i += 256) {
    int row = i >> 6, col = i & 63;
    float v = (row < Kreal) ? W[row * 64 + col] : 0.f;
    int frag = (row >> 5) * 4 + (col >> 4);
    int lane = ((row >> 4) & 1) * 16 + (col & 15);
    Wf[(size_t)((frag * 32 + lane) * 16 + (row & 15))] = (_Float16)v;
  }
}

// Decoder layer-2: W2[64][dout] (dout<=16, zero-pad cols to 16) -> 2 fragments
__device__ __forceinline__ void swizzle_W16(const float* __restrict__ W2,
                                            _Float16* Wf, int dout, int tid) {
  for (int i = tid; i < 64 * 16; i += 256) {
    int row = i >> 4, col = i & 15;
    float v = (col < dout) ? W2[row * dout + col] : 0.f;
    int frag = row >> 5;
    int lane = ((row >> 4) & 1) * 16 + col;
    Wf[(size_t)((frag * 32 + lane) * 16 + (row & 15))] = (_Float16)v;
  }
}

// c[4] = X[16 x 32*KCHUNKS] @ Wf (fragment layout)   (4 N-tiles of 16)
template <int KCHUNKS>
__device__ __forceinline__ void gemm64(const _Float16* X, int ldx,
                                       const _Float16* Wf, int lo, int h,
                                       int lane, v8f c[4]) {
#pragma unroll
  for (int t = 0; t < 4; ++t)
#pragma unroll
    for (int r = 0; r < 8; ++r) c[t][r] = 0.f;
#pragma unroll
  for (int k = 0; k < KCHUNKS; ++k) {
    v16h a = ld_A(X, ldx, k * 32, lo, h);
#pragma unroll
    for (int t = 0; t < 4; ++t) {
      v16h b = ld_Bf(Wf, k * 4 + t, lane);
      c[t] = wmma_f16(a, b, c[t]);
    }
  }
}

// bias + relu, pack f32 accum back to f16 hidden tile H[16][64] in LDS
__device__ __forceinline__ void relu_to_H(const v8f c[4], const float* sB1,
                                          _Float16* H, int lo, int h) {
#pragma unroll
  for (int t = 0; t < 4; ++t) {
    float bb = sB1[t * 16 + lo];
#pragma unroll
    for (int r = 0; r < 8; ++r) {
      float v = c[t][r] + bb;
      H[(r + 8 * h) * LAT + t * 16 + lo] = (_Float16)(v > 0.f ? v : 0.f);
    }
  }
}

// bias + LayerNorm over 64 features/row; row m = r + 8*h spans the 16 lanes of
// one half (cols t*16+lo) -> butterfly reduce with xor masks 1,2,4,8.
__device__ __forceinline__ void bias_ln(v8f c[4], const float* sB2,
                                        const float* sG, const float* sBt,
                                        int lo) {
  float bb[4], gg[4], bt[4];
#pragma unroll
  for (int t = 0; t < 4; ++t) {
    bb[t] = sB2[t * 16 + lo];
    gg[t] = sG[t * 16 + lo];
    bt[t] = sBt[t * 16 + lo];
  }
#pragma unroll
  for (int t = 0; t < 4; ++t)
#pragma unroll
    for (int r = 0; r < 8; ++r) c[t][r] += bb[t];
#pragma unroll
  for (int r = 0; r < 8; ++r) {
    float s1 = 0.f, s2 = 0.f;
#pragma unroll
    for (int t = 0; t < 4; ++t) {
      float v = c[t][r];
      s1 += v;
      s2 += v * v;
    }
#pragma unroll
    for (int m = 1; m < 16; m <<= 1) {
      s1 += __shfl_xor(s1, m, 32);
      s2 += __shfl_xor(s2, m, 32);
    }
    float mu = s1 * (1.f / 64.f);
    float var = s2 * (1.f / 64.f) - mu * mu;
    float rs = rsqrtf(var + LN_EPS);
#pragma unroll
    for (int t = 0; t < 4; ++t) c[t][r] = (c[t][r] - mu) * rs * gg[t] + bt[t];
  }
}

__device__ __forceinline__ void atomAddF(float* p, float v) {
  unsafeAtomicAdd(p, v);  // native global_atomic_add_f32
}

// float4 -> 4 packed f16 (v_cvt_pk_rt_f16_f32 x2) -> one 8B LDS store
__device__ __forceinline__ void cvt_store_h4(float4 d, _Float16* dst) {
  fp16x2 p0 = __builtin_amdgcn_cvt_pkrtz(d.x, d.y);
  fp16x2 p1 = __builtin_amdgcn_cvt_pkrtz(d.z, d.w);
  h4 o;
  o[0] = (_Float16)p0[0];
  o[1] = (_Float16)p0[1];
  o[2] = (_Float16)p1[0];
  o[3] = (_Float16)p1[1];
  *(h4*)dst = o;
}

// ---------------- encoder: feats[count][din] -> out[count][64], din <= 32 ----
__global__ __launch_bounds__(256) void enc_kernel(
    const float* __restrict__ feats, int din, int count,
    const float* __restrict__ W1, const float* __restrict__ B1,
    const float* __restrict__ W2, const float* __restrict__ B2,
    const float* __restrict__ G, const float* __restrict__ Bt,
    float* __restrict__ out) {
  __shared__ __align__(32) _Float16 sW1[32 * LAT];
  __shared__ __align__(32) _Float16 sW2[LAT * LAT];
  __shared__ float sB1[LAT], sB2[LAT], sG[LAT], sBt[LAT];
  __shared__ __align__(32) _Float16 sX[8][16 * 32];
  __shared__ __align__(32) _Float16 sH[8][16 * LAT];

  int tid = threadIdx.x;
  swizzle_W64(W1, sW1, 32, din, tid);
  swizzle_W64(W2, sW2, 64, 64, tid);
  if (tid < LAT) {
    sB1[tid] = B1[tid]; sB2[tid] = B2[tid];
    sG[tid] = G[tid];   sBt[tid] = Bt[tid];
  }
  __syncthreads();

  int w = tid >> 5, lane = tid & 31, lo = lane & 15, h = lane >> 4;
  int row0 = (blockIdx.x * 8 + w) * 16;
  _Float16* X = sX[w];
  _Float16* H = sH[w];

  for (int i = lane; i < 16 * 32; i += 32) {
    int r = i >> 5, k = i & 31;
    float v = 0.f;
    if (k < din && (row0 + r) < count) v = feats[(size_t)(row0 + r) * din + k];
    X[i] = (_Float16)v;
  }

  v8f c[4];
  gemm64<1>(X, 32, sW1, lo, h, lane, c);
  relu_to_H(c, sB1, H, lo, h);
  gemm64<2>(H, LAT, sW2, lo, h, lane, c);
  bias_ln(c, sB2, sG, sBt, lo);

#pragma unroll
  for (int r = 0; r < 8; ++r) {
    int m = row0 + r + 8 * h;
#pragma unroll
    for (int t = 0; t < 4; ++t)
      if (m < count) out[(size_t)m * LAT + t * 16 + lo] = c[t][r];
  }
}

// ------ MP edge update: fused gather + MLP(192->64->64,LN) + residual + scatter
__global__ __launch_bounds__(256) void mp_edge_kernel(
    float* __restrict__ edges, const float* __restrict__ nodes,
    const int* __restrict__ senders, const int* __restrict__ receivers, int E,
    const float* __restrict__ W1, const float* __restrict__ B1,
    const float* __restrict__ W2, const float* __restrict__ B2,
    const float* __restrict__ G, const float* __restrict__ Bt,
    float* __restrict__ aggS, float* __restrict__ aggR) {
  __shared__ __align__(32) _Float16 sW1[192 * LAT];
  __shared__ __align__(32) _Float16 sW2[LAT * LAT];
  __shared__ float sB1[LAT], sB2[LAT], sG[LAT], sBt[LAT];
  __shared__ __align__(32) _Float16 sX[8][16 * 192];

  int tid = threadIdx.x;
  swizzle_W64(W1, sW1, 192, 192, tid);
  swizzle_W64(W2, sW2, 64, 64, tid);
  if (tid < LAT) {
    sB1[tid] = B1[tid]; sB2[tid] = B2[tid];
    sG[tid] = G[tid];   sBt[tid] = Bt[tid];
  }
  __syncthreads();

  int w = tid >> 5, lane = tid & 31, lo = lane & 15, h = lane >> 4;
  int e0 = (blockIdx.x * 8 + w) * 16;
  _Float16* X = sX[w];
  _Float16* H = sX[w];  // layer-1 input dead before hidden written: alias OK

  int el = e0 + lo;
  int ec = el < E ? el : (E - 1);
  int sv = senders[ec];
  int rv = receivers[ec];

  // X = [edge_lat | nodes[sender] | nodes[receiver]] as f16, float4 loads
#pragma unroll 4
  for (int it = 0; it < 24; ++it) {
    int li = lane + 32 * it;  // 0..767 (16 rows x 48 float4)
    int row = li / 48, q = li - row * 48;
    int seg = q >> 4, c4 = q & 15;
    int e = e0 + row;
    bool valid = e < E;
    int sm = __shfl(sv, row, 32);
    int rm = __shfl(rv, row, 32);
    const float* src;
    if (seg == 0)      src = edges + (size_t)(valid ? e : 0) * LAT;
    else if (seg == 1) src = nodes + (size_t)sm * LAT;
    else               src = nodes + (size_t)rm * LAT;
    float4 d = valid ? *(const float4*)(src + c4 * 4)
                     : make_float4(0.f, 0.f, 0.f, 0.f);
    cvt_store_h4(d, X + row * 192 + q * 4);
  }

  v8f c[4];
  gemm64<6>(X, 192, sW1, lo, h, lane, c);
  relu_to_H(c, sB1, H, lo, h);
  gemm64<2>(H, LAT, sW2, lo, h, lane, c);
  bias_ln(c, sB2, sG, sBt, lo);

  // residual into edges + scatter-add new_edges into sender/receiver aggs
#pragma unroll
  for (int r = 0; r < 8; ++r) {
    int m = r + 8 * h;
    int e = e0 + m;
    int sm = __shfl(sv, m, 32);
    int rm = __shfl(rv, m, 32);
#pragma unroll
    for (int t = 0; t < 4; ++t) {
      int col = t * 16 + lo;
      float v = c[t][r];
      if (e < E) {
        size_t off = (size_t)e * LAT + col;
        edges[off] = edges[off] + v;
        atomAddF(&aggS[(size_t)sm * LAT + col], v);
        atomAddF(&aggR[(size_t)rm * LAT + col], v);
      }
    }
  }
}

// ------ MP node update: MLP([nodes|aggS|aggR]) + residual --------------------
__global__ __launch_bounds__(256) void mp_node_kernel(
    float* __restrict__ nodes, const float* __restrict__ aggS,
    const float* __restrict__ aggR, int Ncnt,
    const float* __restrict__ W1, const float* __restrict__ B1,
    const float* __restrict__ W2, const float* __restrict__ B2,
    const float* __restrict__ G, const float* __restrict__ Bt) {
  __shared__ __align__(32) _Float16 sW1[192 * LAT];
  __shared__ __align__(32) _Float16 sW2[LAT * LAT];
  __shared__ float sB1[LAT], sB2[LAT], sG[LAT], sBt[LAT];
  __shared__ __align__(32) _Float16 sX[8][16 * 192];

  int tid = threadIdx.x;
  swizzle_W64(W1, sW1, 192, 192, tid);
  swizzle_W64(W2, sW2, 64, 64, tid);
  if (tid < LAT) {
    sB1[tid] = B1[tid]; sB2[tid] = B2[tid];
    sG[tid] = G[tid];   sBt[tid] = Bt[tid];
  }
  __syncthreads();

  int w = tid >> 5, lane = tid & 31, lo = lane & 15, h = lane >> 4;
  int n0 = (blockIdx.x * 8 + w) * 16;
  _Float16* X = sX[w];
  _Float16* H = sX[w];

#pragma unroll 4
  for (int it = 0; it < 24; ++it) {
    int li = lane + 32 * it;
    int row = li / 48, q = li - row * 48;
    int seg = q >> 4, c4 = q & 15;
    int n = n0 + row;
    bool valid = n < Ncnt;
    const float* base = (seg == 0) ? nodes : ((seg == 1) ? aggS : aggR);
    const float* src = base + (size_t)(valid ? n : 0) * LAT;
    float4 d = valid ? *(const float4*)(src + c4 * 4)
                     : make_float4(0.f, 0.f, 0.f, 0.f);
    cvt_store_h4(d, X + row * 192 + q * 4);
  }

  v8f c[4];
  gemm64<6>(X, 192, sW1, lo, h, lane, c);
  relu_to_H(c, sB1, H, lo, h);
  gemm64<2>(H, LAT, sW2, lo, h, lane, c);
  bias_ln(c, sB2, sG, sBt, lo);

#pragma unroll
  for (int r = 0; r < 8; ++r) {
    int n = n0 + r + 8 * h;
#pragma unroll
    for (int t = 0; t < 4; ++t) {
      if (n < Ncnt) {
        size_t off = (size_t)n * LAT + t * 16 + lo;
        nodes[off] = nodes[off] + c[t][r];
      }
    }
  }
}

// ------ decoder: lat[count][64] -> out[count][dout], dout <= 16, no LN ------
__global__ __launch_bounds__(256) void dec_kernel(
    const float* __restrict__ lat, int count, int dout,
    const float* __restrict__ W1, const float* __restrict__ B1,
    const float* __restrict__ W2, const float* __restrict__ B2,
    float* __restrict__ out) {
  __shared__ __align__(32) _Float16 sW1[LAT * LAT];
  __shared__ __align__(32) _Float16 sW2[LAT * 16];
  __shared__ float sB1[LAT], sB2[16];
  __shared__ __align__(32) _Float16 sX[8][16 * LAT];

  int tid = threadIdx.x;
  swizzle_W64(W1, sW1, 64, 64, tid);
  swizzle_W16(W2, sW2, dout, tid);
  if (tid < LAT) sB1[tid] = B1[tid];
  if (tid < 16) sB2[tid] = (tid < dout) ? B2[tid] : 0.f;
  __syncthreads();

  int w = tid >> 5, lane = tid & 31, lo = lane & 15, h = lane >> 4;
  int row0 = (blockIdx.x * 8 + w) * 16;
  _Float16* X = sX[w];
  _Float16* H = sX[w];

  for (int i = lane; i < 16 * 16; i += 32) {  // 16 rows x 16 float4
    int r = i >> 4, c4 = i & 15;
    float4 d = ((row0 + r) < count)
                   ? *(const float4*)(lat + (size_t)(row0 + r) * LAT + c4 * 4)
                   : make_float4(0.f, 0.f, 0.f, 0.f);
    cvt_store_h4(d, X + r * LAT + c4 * 4);
  }

  v8f c[4];
  gemm64<2>(X, LAT, sW1, lo, h, lane, c);
  relu_to_H(c, sB1, H, lo, h);

  v8f c2;
#pragma unroll
  for (int r = 0; r < 8; ++r) c2[r] = 0.f;
#pragma unroll
  for (int k = 0; k < 2; ++k) {
    v16h a = ld_A(H, LAT, k * 32, lo, h);
    v16h b = ld_Bf(sW2, k, lane);
    c2 = wmma_f16(a, b, c2);
  }

  if (lo < dout) {
#pragma unroll
    for (int r = 0; r < 8; ++r) {
      int m = row0 + r + 8 * h;
      if (m < count) out[(size_t)m * dout + lo] = c2[r] + sB2[lo];
    }
  }
}

extern "C" void kernel_launch(void* const* d_in, const int* in_sizes, int n_in,
                              void* d_out, int out_size, void* d_ws,
                              size_t ws_size, hipStream_t stream) {
  const int N = in_sizes[0] / 16;  // node_feats [N,16]
  const int E = in_sizes[2];       // senders [E]
  const int E0 = E / 2;

  const float* node_feats = (const float*)d_in[0];
  const float* edge_feats = (const float*)d_in[1];
  const int* senders = (const int*)d_in[2];
  const int* receivers = (const int*)d_in[3];

  // params in dict insertion order:
  // enc_e0(6) enc_e1(6) enc_n(6) mp_e(6) mp_n(6) dec_e0(4) dec_e1(4) dec_n(4)
  const float* P[42];
  for (int i = 0; i < 42; ++i) P[i] = (const float*)d_in[4 + i];
  const float* const* enc_e0 = P + 0;
  const float* const* enc_e1 = P + 6;
  const float* const* enc_n  = P + 12;
  const float* const* mp_e   = P + 18;
  const float* const* mp_n   = P + 24;
  const float* const* dec_e0 = P + 30;
  const float* const* dec_e1 = P + 34;
  const float* const* dec_n  = P + 38;

  float* ws = (float*)d_ws;
  float* nodes_lat = ws;                      // N*64
  float* aggS = nodes_lat + (size_t)N * LAT;  // N*64
  float* aggR = aggS + (size_t)N * LAT;       // N*64
  float* edges_lat = aggR + (size_t)N * LAT;  // E*64

  dim3 blk(256);
  auto nblocks = [](int cnt) { return dim3((unsigned)((cnt + 127) / 128)); };

  enc_kernel<<<nblocks(E0), blk, 0, stream>>>(
      edge_feats, 8, E0, enc_e0[0], enc_e0[1], enc_e0[2], enc_e0[3], enc_e0[4],
      enc_e0[5], edges_lat);
  enc_kernel<<<nblocks(E - E0), blk, 0, stream>>>(
      edge_feats + (size_t)E0 * 8, 8, E - E0, enc_e1[0], enc_e1[1], enc_e1[2],
      enc_e1[3], enc_e1[4], enc_e1[5], edges_lat + (size_t)E0 * LAT);
  enc_kernel<<<nblocks(N), blk, 0, stream>>>(
      node_feats, 16, N, enc_n[0], enc_n[1], enc_n[2], enc_n[3], enc_n[4],
      enc_n[5], nodes_lat);

  for (int step = 0; step < 2; ++step) {
    (void)hipMemsetAsync(aggS, 0, (size_t)N * LAT * 2 * sizeof(float), stream);
    mp_edge_kernel<<<nblocks(E), blk, 0, stream>>>(
        edges_lat, nodes_lat, senders, receivers, E, mp_e[0], mp_e[1], mp_e[2],
        mp_e[3], mp_e[4], mp_e[5], aggS, aggR);
    mp_node_kernel<<<nblocks(N), blk, 0, stream>>>(
        nodes_lat, aggS, aggR, N, mp_n[0], mp_n[1], mp_n[2], mp_n[3], mp_n[4],
        mp_n[5]);
  }

  float* out = (float*)d_out;
  dec_kernel<<<nblocks(N), blk, 0, stream>>>(nodes_lat, N, 3, dec_n[0],
                                             dec_n[1], dec_n[2], dec_n[3], out);
  dec_kernel<<<nblocks(E0), blk, 0, stream>>>(edges_lat, E0, 1, dec_e0[0],
                                              dec_e0[1], dec_e0[2], dec_e0[3],
                                              out + (size_t)N * 3);
  dec_kernel<<<nblocks(E - E0), blk, 0, stream>>>(
      edges_lat + (size_t)E0 * LAT, E - E0, 1, dec_e1[0], dec_e1[1], dec_e1[2],
      dec_e1[3], out + (size_t)N * 3 + E0);
}
